// ZorroTransformer_16415365005316
// MI455X (gfx1250) — matrix operand order repaired and checked by reading, not compiled, not run on hardware
//
#include <hip/hip_runtime.h>
#include <math.h>

typedef __attribute__((ext_vector_type(16))) __bf16    v16bf;
typedef __attribute__((ext_vector_type(8)))  float     v8f;
typedef __attribute__((ext_vector_type(4)))  unsigned  v4u;
typedef __attribute__((ext_vector_type(4)))  int       v4i;

#if __has_builtin(__builtin_amdgcn_global_load_async_to_lds_b128) && \
    __has_builtin(__builtin_amdgcn_s_wait_asynccnt)
#define HAVE_ASYNC 1
typedef __attribute__((address_space(1))) v4i* g_v4i_p;   // global int4*
typedef __attribute__((address_space(3))) v4i* l_v4i_p;   // LDS int4*
#else
#define HAVE_ASYNC 0
#endif

namespace zc {
constexpr int B = 2, S = 976, D = 1024, H = 16, DH = 64, L = 4, FF = 4096;
constexpr int SP = 992;                               // padded K-stride (probs, vT)
constexpr long TOK_OUT = (long)B * S * D;             // floats of token outputs
constexpr long ATTN_PER_LAYER = (long)B * H * S * S;  // floats per layer attn
constexpr int SLACK = 128;                            // row slack for unguarded loads
}
using namespace zc;

// ---------------- Zorro mask ----------------
__device__ __forceinline__ int tok_type(int i) {
  return (i < 512) ? 0 : (i < 768) ? 1 : (i < 896) ? 2 : (i < 960) ? 3 : 4;
}
__device__ __forceinline__ bool zmask(int i, int j) {
  int ti = tok_type(i), tj = tok_type(j);
  return (ti == tj) || (ti == 4) || (ti == 3 && j < 960);
}

// ---------------- concat streams -> tokens ----------------
__global__ void concat_kernel(const float* __restrict__ r0, const float* __restrict__ r1,
                              const float* __restrict__ r2, const float* __restrict__ r3,
                              const float* __restrict__ r4, float* __restrict__ tok) {
  long idx = (long)blockIdx.x * blockDim.x + threadIdx.x;
  long total = (long)B * S * D;
  if (idx >= total) return;
  int d = (int)(idx % D);
  long t = idx / D;
  int s = (int)(t % S);
  int b = (int)(t / S);
  const float* src; int sl, cnt;
  if (s < 512)      { src = r0; sl = s;       cnt = 512; }
  else if (s < 768) { src = r1; sl = s - 512; cnt = 256; }
  else if (s < 896) { src = r2; sl = s - 768; cnt = 128; }
  else if (s < 960) { src = r3; sl = s - 896; cnt = 64;  }
  else              { src = r4; sl = s - 960; cnt = 16;  }
  tok[idx] = src[((long)b * cnt + sl) * D + d];
}

// ---------------- transpose-convert: f32 in[K][N] -> bf16 out[N][K] ----------------
__global__ __launch_bounds__(256)
void cvtT_kernel(const float* __restrict__ in, __bf16* __restrict__ out, int K, int N) {
  __shared__ __bf16 t[32][33];
  int tx = threadIdx.x, ty = threadIdx.y;
  int k0 = blockIdx.x * 32, n0 = blockIdx.y * 32;
#pragma unroll
  for (int j = 0; j < 4; ++j) {
    int k = k0 + ty + 8 * j;
    t[tx][ty + 8 * j] = (__bf16)in[(long)k * N + n0 + tx];  // t[n_local][k_local]
  }
  __syncthreads();
#pragma unroll
  for (int j = 0; j < 4; ++j) {
    int n = n0 + ty + 8 * j;
    out[(long)n * K + k0 + tx] = t[ty + 8 * j][tx];
  }
}

// ---------------- v transpose: kvb v-part [S][1024] -> vT[1024][SP], zero K-tail ----------------
__global__ __launch_bounds__(256)
void vt_kernel(const __bf16* __restrict__ kvb, __bf16* __restrict__ vT) {
  __shared__ __bf16 t[32][33];
  int tx = threadIdx.x, ty = threadIdx.y;
  int b = blockIdx.z;
  int s0 = blockIdx.x * 32, n0 = blockIdx.y * 32;
#pragma unroll
  for (int j = 0; j < 4; ++j) {
    int s = s0 + ty + 8 * j;
    __bf16 val = (__bf16)0.f;
    if (s < S) val = kvb[((long)b * S + s) * (2 * D) + D + n0 + tx];
    t[tx][ty + 8 * j] = val;  // t[n_local][s_local]
  }
  __syncthreads();
#pragma unroll
  for (int j = 0; j < 4; ++j) {
    int n = n0 + ty + 8 * j;
    vT[((long)b * D + n) * SP + s0 + tx] = t[ty + 8 * j][tx];
  }
}

// ---------------- LayerNorm -> bf16 (D=1024, 256 threads/row) ----------------
__global__ __launch_bounds__(256)
void ln_bf16_kernel(const float* __restrict__ x, const float* __restrict__ g,
                    __bf16* __restrict__ y) {
  __shared__ float red[256];
  long row = blockIdx.x;
  const float* xr = x + row * D;
  int tid = threadIdx.x;
  float v[4], s = 0.f, s2 = 0.f;
#pragma unroll
  for (int i = 0; i < 4; ++i) { v[i] = xr[tid + (i << 8)]; s += v[i]; s2 += v[i] * v[i]; }
  red[tid] = s; __syncthreads();
  for (int off = 128; off > 0; off >>= 1) { if (tid < off) red[tid] += red[tid + off]; __syncthreads(); }
  float mu = red[0] * (1.f / D);
  __syncthreads();
  red[tid] = s2; __syncthreads();
  for (int off = 128; off > 0; off >>= 1) { if (tid < off) red[tid] += red[tid + off]; __syncthreads(); }
  float var = red[0] * (1.f / D) - mu * mu;
  float inv = rsqrtf(var + 1e-5f);
  __bf16* yr = y + row * D;
#pragma unroll
  for (int i = 0; i < 4; ++i) { int d = tid + (i << 8); yr[d] = (__bf16)((v[i] - mu) * inv * g[d]); }
}

// ---------------- Final LayerNorm -> split f32 outputs ----------------
__global__ __launch_bounds__(256)
void ln_final_kernel(const float* __restrict__ x, const float* __restrict__ g,
                     float* __restrict__ out) {
  __shared__ float red[256];
  long row = blockIdx.x;
  const float* xr = x + row * D;
  int tid = threadIdx.x;
  float v[4], s = 0.f, s2 = 0.f;
#pragma unroll
  for (int i = 0; i < 4; ++i) { v[i] = xr[tid + (i << 8)]; s += v[i]; s2 += v[i] * v[i]; }
  red[tid] = s; __syncthreads();
  for (int off = 128; off > 0; off >>= 1) { if (tid < off) red[tid] += red[tid + off]; __syncthreads(); }
  float mu = red[0] * (1.f / D);
  __syncthreads();
  red[tid] = s2; __syncthreads();
  for (int off = 128; off > 0; off >>= 1) { if (tid < off) red[tid] += red[tid + off]; __syncthreads(); }
  float var = red[0] * (1.f / D) - mu * mu;
  float inv = rsqrtf(var + 1e-5f);
  int s_idx = (int)(row % S), b = (int)(row / S);
  long obase; int sl, cnt;
  if (s_idx < 512)      { obase = 0;                          sl = s_idx;       cnt = 512; }
  else if (s_idx < 768) { obase = (long)B * 512 * D;          sl = s_idx - 512; cnt = 256; }
  else if (s_idx < 896) { obase = (long)B * (512 + 256) * D;  sl = s_idx - 768; cnt = 128; }
  else if (s_idx < 960) { obase = (long)B * (512 + 256 + 128) * D;      sl = s_idx - 896; cnt = 64; }
  else                  { obase = (long)B * (512 + 256 + 128 + 64) * D; sl = s_idx - 960; cnt = 16; }
  float* dst = out + obase + ((long)b * cnt + sl) * D;
#pragma unroll
  for (int i = 0; i < 4; ++i) { int d = tid + (i << 8); dst[d] = (v[i] - mu) * inv * g[d]; }
}

// ---------------- row softmax (len 976), in-place f32 + padded bf16 copy ----------------
__global__ __launch_bounds__(256)
void softmax_kernel(float* __restrict__ p, __bf16* __restrict__ pb) {
  __shared__ float red[256];
  long row = blockIdx.x;
  float* pr = p + row * S;
  __bf16* pbr = pb + row * SP;
  int tid = threadIdx.x;
  float v[4], mx = -3.402823466e38f;
#pragma unroll
  for (int i = 0; i < 4; ++i) {
    int j = tid + (i << 8);
    v[i] = (j < S) ? pr[j] : -3.402823466e38f;
    mx = fmaxf(mx, v[i]);
  }
  red[tid] = mx; __syncthreads();
  for (int off = 128; off > 0; off >>= 1) { if (tid < off) red[tid] = fmaxf(red[tid], red[tid + off]); __syncthreads(); }
  mx = red[0]; __syncthreads();
  float sum = 0.f;
#pragma unroll
  for (int i = 0; i < 4; ++i) {
    int j = tid + (i << 8);
    float e = (j < S) ? __expf(v[i] - mx) : 0.f;
    v[i] = e; sum += e;
  }
  red[tid] = sum; __syncthreads();
  for (int off = 128; off > 0; off >>= 1) { if (tid < off) red[tid] += red[tid + off]; __syncthreads(); }
  float inv = 1.f / red[0];
#pragma unroll
  for (int i = 0; i < 4; ++i) {
    int j = tid + (i << 8);
    if (j < S)       { float q = v[i] * inv; pr[j] = q; pbr[j] = (__bf16)q; }
    else if (j < SP) { pbr[j] = (__bf16)0.f; }  // zero K-padding for attn@v
  }
}

// ---------------- bf16 WMMA GEMM, 128x64x32 tiles, double-buffered async LDS ----------------
// B operand is ALWAYS transposed layout: element (k,n) = Bm[n*ldb + k].
// 256 threads = 8 waves (4M x 2N); each wave computes 32x32 via four
// v_wmma_f32_16x16x32_bf16 per K step. Loads unguarded (callers give slack);
// K tails (attn@v) neutralized by zero-padded probs (A) and vT (B).
template <bool OUTB, bool RES, bool MS, bool GELU_>
__global__ __launch_bounds__(256)
void gemm_wmma(const __bf16* __restrict__ A, int lda, long aO, long aI,
               const __bf16* __restrict__ Bm, int ldb, long bO, long bI,
               float* __restrict__ Cf, __bf16* __restrict__ Cb, int ldc, long cO, long cI,
               int M, int N, int K, float scale) {
  __shared__ __align__(16) __bf16 As[2][128][40];
  __shared__ __align__(16) __bf16 Bs[2][64][40];  // column-major: Bs[n][k]
  const int tid = threadIdx.x;
  const int wave = tid >> 5, lane = tid & 31;
  const int m32 = wave >> 1, nHalf = wave & 1;
  const int lm = lane & 15, lh = lane >> 4;
  const int z = blockIdx.z;
  const int zb = z >> 4, zh = z & 15;
  A  += (long)zb * aO + (long)zh * aI;
  Bm += (long)zb * bO + (long)zh * bI;
  if (OUTB) Cb += (long)zb * cO + (long)zh * cI;
  else      Cf += (long)zb * cO + (long)zh * cI;
  const int mBase = blockIdx.y * 128;
  const int nBase = blockIdx.x * 64;
  v8f c00 = {}, c01 = {}, c10 = {}, c11 = {};
  const int kTiles = (K + 31) >> 5;

  auto stage = [&](int kt_, int buf_) {
    const int k0 = kt_ << 5;
#pragma unroll
    for (int i = 0; i < 2; ++i) {          // A tile: 128x32, 2 x b128 per thread
      int linear = tid + (i << 8);
      int r = linear >> 2, c8 = (linear & 3) << 3;
      const __bf16* src = A + (long)(mBase + r) * lda + k0 + c8;
      __bf16* dst = &As[buf_][r][c8];
#if HAVE_ASYNC
      __builtin_amdgcn_global_load_async_to_lds_b128((g_v4i_p)src, (l_v4i_p)dst, 0, 0);
#else
      *(v4u*)dst = *(const v4u*)src;
#endif
    }
    {                                      // B tile: 64n x 32k, 1 x b128 per thread
      int n = tid >> 2, k8 = (tid & 3) << 3;
      const __bf16* src = Bm + (long)(nBase + n) * ldb + k0 + k8;
      __bf16* dst = &Bs[buf_][n][k8];
#if HAVE_ASYNC
      __builtin_amdgcn_global_load_async_to_lds_b128((g_v4i_p)src, (l_v4i_p)dst, 0, 0);
#else
      *(v4u*)dst = *(const v4u*)src;
#endif
    }
  };

  stage(0, 0);
  int buf = 0;
  for (int kt = 0; kt < kTiles; ++kt) {
    if (kt + 1 < kTiles) {
      stage(kt + 1, buf ^ 1);   // prefetch next tile into other buffer
#if HAVE_ASYNC
      __builtin_amdgcn_s_wait_asynccnt(3);   // current tile's 3 asyncs done
#endif
    } else {
#if HAVE_ASYNC
      __builtin_amdgcn_s_wait_asynccnt(0);
#endif
    }
    __syncthreads();
    // ---- fragments (all b128 LDS loads) ----
    struct U32x8 { v4u lo, hi; };
    const int arow0 = (m32 << 5) + lm;
    U32x8 ta0{ *(const v4u*)(&As[buf][arow0][lh << 3]),      *(const v4u*)(&As[buf][arow0][16 + (lh << 3)]) };
    U32x8 ta1{ *(const v4u*)(&As[buf][arow0 + 16][lh << 3]), *(const v4u*)(&As[buf][arow0 + 16][16 + (lh << 3)]) };
    const int col0 = (nHalf << 5) + lm;
    U32x8 tb0{ *(const v4u*)(&Bs[buf][col0][lh << 3]),      *(const v4u*)(&Bs[buf][col0][16 + (lh << 3)]) };
    U32x8 tb1{ *(const v4u*)(&Bs[buf][col0 + 16][lh << 3]), *(const v4u*)(&Bs[buf][col0 + 16][16 + (lh << 3)]) };
    v16bf a0 = __builtin_bit_cast(v16bf, ta0);
    v16bf a1 = __builtin_bit_cast(v16bf, ta1);
    v16bf b0 = __builtin_bit_cast(v16bf, tb0);
    v16bf b1 = __builtin_bit_cast(v16bf, tb1);
    c00 = __builtin_amdgcn_wmma_f32_16x16x32_bf16(false, a0, false, b0, (short)0, c00, false, false);
    c01 = __builtin_amdgcn_wmma_f32_16x16x32_bf16(false, a0, false, b1, (short)0, c01, false, false);
    c10 = __builtin_amdgcn_wmma_f32_16x16x32_bf16(false, a1, false, b0, (short)0, c10, false, false);
    c11 = __builtin_amdgcn_wmma_f32_16x16x32_bf16(false, a1, false, b1, (short)0, c11, false, false);
    __syncthreads();
    buf ^= 1;
  }
  // ---- store (C layout: lane n=lm, rows lh*8 + r) ----
#pragma unroll
  for (int sm = 0; sm < 2; ++sm) {
#pragma unroll
    for (int sn = 0; sn < 2; ++sn) {
      v8f cc = sm ? (sn ? c11 : c10) : (sn ? c01 : c00);
      int col = nBase + (nHalf << 5) + (sn << 4) + lm;
      if (col < N) {
        int rowb = mBase + (m32 << 5) + (sm << 4) + (lh << 3);
#pragma unroll
        for (int r = 0; r < 8; ++r) {
          int row = rowb + r;
          if (row < M) {
            float v = cc[r];
            if (MS) {
              v *= scale;
              if (!zmask(row, col)) v = -3.402823466e38f;
            }
            if (GELU_) v = 0.5f * v * (1.0f + erff(v * 0.70710678118f));
            if (RES) v += Cf[(long)row * ldc + col];
            if (OUTB) Cb[(long)row * ldc + col] = (__bf16)v;
            else      Cf[(long)row * ldc + col] = v;
          }
        }
      }
    }
  }
}

// ---------------- host orchestration ----------------
extern "C" void kernel_launch(void* const* d_in, const int* in_sizes, int n_in,
                              void* d_out, int out_size, void* d_ws, size_t ws_size,
                              hipStream_t stream) {
  (void)in_sizes; (void)n_in; (void)out_size; (void)ws_size;
  const float* rna        = (const float*)d_in[0];
  const float* roi        = (const float*)d_in[1];
  const float* structure  = (const float*)d_in[2];
  const float* expression = (const float*)d_in[3];
  const float* fusion     = (const float*)d_in[4];
  const float* qw    = (const float*)d_in[5];
  const float* kvw   = (const float*)d_in[6];
  const float* ow    = (const float*)d_in[7];
  const float* ag    = (const float*)d_in[8];
  const float* fg    = (const float*)d_in[9];
  const float* w1    = (const float*)d_in[10];
  const float* w2    = (const float*)d_in[11];
  const float* gamma = (const float*)d_in[12];
  float* out = (float*)d_out;

  char* ws = (char*)d_ws;
  size_t off = 0;
  auto carve = [&](size_t bytes) -> char* {
    char* p = ws + off;
    off = (off + bytes + 255) & ~(size_t)255;
    return p;
  };
  float*  tokens = (float*) carve((size_t)B * S * D * 4);
  __bf16* xb     = (__bf16*)carve((size_t)(B * S + SLACK) * D * 2);
  __bf16* qb     = (__bf16*)carve((size_t)(B * S + SLACK) * D * 2);
  __bf16* kvb    = (__bf16*)carve((size_t)(B * S + SLACK) * 2 * D * 2);
  __bf16* ctx    = (__bf16*)carve((size_t)(B * S + SLACK) * D * 2);
  __bf16* hb     = (__bf16*)carve((size_t)(B * S + SLACK) * FF * 2);
  __bf16* pb     = (__bf16*)carve(((size_t)B * H * S + SLACK) * SP * 2);
  __bf16* vT     = (__bf16*)carve((size_t)B * D * SP * 2);
  __bf16* wqT    = (__bf16*)carve((size_t)D * D * 2);        // [N=1024][K=1024]
  __bf16* wkvT   = (__bf16*)carve((size_t)2 * D * D * 2);    // [2048][1024]
  __bf16* woT    = (__bf16*)carve((size_t)D * D * 2);        // [1024][1024]
  __bf16* w1T    = (__bf16*)carve((size_t)D * FF * 2);       // [4096][1024]
  __bf16* w2T    = (__bf16*)carve((size_t)FF * D * 2);       // [1024][4096]

  dim3 t256(256), t32x8(32, 8);
  long total = (long)B * S * D;
  concat_kernel<<<dim3((unsigned)((total + 255) / 256)), t256, 0, stream>>>(
      rna, roi, structure, expression, fusion, tokens);

  const int Mflat = B * S;                 // 1952
  const int gM  = (Mflat + 127) / 128;     // 16
  const int gMS = (S + 127) / 128;         // 8

  for (int l = 0; l < L; ++l) {
    cvtT_kernel<<<dim3(D / 32, D / 32), t32x8, 0, stream>>>(qw + (long)l * D * D, wqT, D, D);
    cvtT_kernel<<<dim3(D / 32, 2 * D / 32), t32x8, 0, stream>>>(kvw + (long)l * D * 2 * D, wkvT, D, 2 * D);
    cvtT_kernel<<<dim3(D / 32, D / 32), t32x8, 0, stream>>>(ow + (long)l * D * D, woT, D, D);
    cvtT_kernel<<<dim3(D / 32, FF / 32), t32x8, 0, stream>>>(w1 + (long)l * D * FF, w1T, D, FF);
    cvtT_kernel<<<dim3(FF / 32, D / 32), t32x8, 0, stream>>>(w2 + (long)l * FF * D, w2T, FF, D);

    ln_bf16_kernel<<<Mflat, t256, 0, stream>>>(tokens, ag + (long)l * D, xb);

    // q = ln(x) @ qw  -> bf16
    gemm_wmma<true, false, false, false><<<dim3(D / 64, gM, 1), t256, 0, stream>>>(
        xb, D, 0, 0, wqT, D, 0, 0, nullptr, qb, D, 0, 0, Mflat, D, D, 1.f);
    // kv = ln(x) @ kvw -> bf16 (k cols [0,1024), v cols [1024,2048))
    gemm_wmma<true, false, false, false><<<dim3(2 * D / 64, gM, 1), t256, 0, stream>>>(
        xb, D, 0, 0, wkvT, D, 0, 0, nullptr, kvb, 2 * D, 0, 0, Mflat, 2 * D, D, 1.f);
    // vT = transpose(v) with zeroed K-tail
    vt_kernel<<<dim3(SP / 32, D / 32, B), t32x8, 0, stream>>>(kvb, vT);

    float* attnL = out + TOK_OUT + (long)l * ATTN_PER_LAYER;
    // sim = scale * q @ k^T with Zorro mask, per (b,h), logits -> d_out
    gemm_wmma<false, false, true, false><<<dim3((S + 63) / 64, gMS, B * H), t256, 0, stream>>>(
        qb, D, (long)S * D, DH,
        kvb, 2 * D, (long)S * 2 * D, DH,
        attnL, nullptr, S, (long)H * S * S, (long)S * S,
        S, S, DH, 0.125f);
    softmax_kernel<<<B * H * S, t256, 0, stream>>>(attnL, pb);
    // ctx = attn @ v, per (b,h) -> bf16 (B,S,H*DH)
    gemm_wmma<true, false, false, false><<<dim3(1, gMS, B * H), t256, 0, stream>>>(
        pb, SP, (long)H * S * SP, (long)S * SP,
        vT, SP, (long)D * SP, (long)DH * SP,
        nullptr, ctx, D, (long)S * D, DH,
        S, DH, S, 1.f);
    // tokens += ctx @ ow  (fused residual, f32)
    gemm_wmma<false, true, false, false><<<dim3(D / 64, gM, 1), t256, 0, stream>>>(
        ctx, D, 0, 0, woT, D, 0, 0, tokens, nullptr, D, 0, 0, Mflat, D, D, 1.f);

    ln_bf16_kernel<<<Mflat, t256, 0, stream>>>(tokens, fg + (long)l * D, xb);
    // h = gelu(ln @ w1) -> bf16
    gemm_wmma<true, false, false, true><<<dim3(FF / 64, gM, 1), t256, 0, stream>>>(
        xb, D, 0, 0, w1T, D, 0, 0, nullptr, hb, FF, 0, 0, Mflat, FF, D, 1.f);
    // tokens += h @ w2  (fused residual, f32)
    gemm_wmma<false, true, false, false><<<dim3(D / 64, gM, 1), t256, 0, stream>>>(
        hb, FF, 0, 0, w2T, FF, 0, 0, tokens, nullptr, D, 0, 0, Mflat, D, D, 1.f);
  }

  ln_final_kernel<<<Mflat, t256, 0, stream>>>(tokens, gamma, out);
}
